// UnsupEmbedding_RoIHead_16904991277698
// MI455X (gfx1250) — matrix operand order repaired and checked by reading, not compiled, not run on hardware
//
#include <hip/hip_runtime.h>
#include <hip/hip_bf16.h>
#include <math.h>

// ---------------------------------------------------------------------------
// Problem constants (from reference): feats [N,C,H,W], centroids [K,C,H,W]
// ---------------------------------------------------------------------------
#define N_SAMP 256
#define C_CH   2048
#define HW     196          // 14*14
#define HW4    49           // 196 floats = 49 float4
#define K_CLS  7

typedef _Float16 v16h __attribute__((ext_vector_type(16)));
typedef _Float16 v8h  __attribute__((ext_vector_type(8)));
typedef float    v8f  __attribute__((ext_vector_type(8)));
typedef int      v4i  __attribute__((ext_vector_type(4)));

typedef __attribute__((address_space(1))) v4i* gbl_v4i_p;  // global int4*
typedef __attribute__((address_space(3))) v4i* lds_v4i_p;  // LDS int4*

// ---------------------------------------------------------------------------
// gfx1250 async global->LDS copy (16B per lane), ASYNCcnt-tracked.
// Builtin signature (from probe): (v4i addrspace(1)*, v4i addrspace(3)*,
// imm offset, imm cpol).  Falls back to a plain vector copy if absent.
// ---------------------------------------------------------------------------
#if __has_builtin(__builtin_amdgcn_global_load_async_to_lds_b128)
#define HAVE_ASYNC_LDS 1
#else
#define HAVE_ASYNC_LDS 0
#endif

__device__ __forceinline__ void async_copy16(const _Float16* gsrc, _Float16* ldst) {
#if HAVE_ASYNC_LDS
    __builtin_amdgcn_global_load_async_to_lds_b128(
        (gbl_v4i_p)gsrc, (lds_v4i_p)ldst, /*imm_offset=*/0, /*cpol=*/0);
#else
    *(v8h*)ldst = *(const v8h*)gsrc;
#endif
}

__device__ __forceinline__ void async_copy_wait() {
#if HAVE_ASYNC_LDS
#if __has_builtin(__builtin_amdgcn_s_wait_asynccnt)
    __builtin_amdgcn_s_wait_asynccnt(0);
#else
    asm volatile("s_wait_asynccnt 0" ::: "memory");
#endif
#endif
}

// ---------------------------------------------------------------------------
// Kernel 0: one-time f32 -> f16 conversion of W_sel (L2-resident, 16MB read).
// Each thread converts 8 elements (two float4 loads -> one 16B f16 store).
// ---------------------------------------------------------------------------
__global__ __launch_bounds__(256) void cvt_f16_kernel(const float* __restrict__ src,
                                                      _Float16* __restrict__ dst,
                                                      int n8) {
    const int i = blockIdx.x * 256 + threadIdx.x;   // unit: 8 elements
    if (i >= n8) return;
    const float4 f0 = ((const float4*)src)[(size_t)i * 2 + 0];
    const float4 f1 = ((const float4*)src)[(size_t)i * 2 + 1];
    v8h h;
    h[0] = (_Float16)f0.x; h[1] = (_Float16)f0.y;
    h[2] = (_Float16)f0.z; h[3] = (_Float16)f0.w;
    h[4] = (_Float16)f1.x; h[5] = (_Float16)f1.y;
    h[6] = (_Float16)f1.z; h[7] = (_Float16)f1.w;
    ((v8h*)dst)[i] = h;
}

// ---------------------------------------------------------------------------
// Kernel 1: spatial mean per (n,c), emitted directly as f16 (GEMM A operand).
// One wave32 per (n,c); lanes read coalesced 128B segments, shfl-xor reduce.
// ---------------------------------------------------------------------------
__global__ __launch_bounds__(256) void pool_kernel(const float* __restrict__ feats,
                                                   _Float16* __restrict__ pooledh) {
    const int gw   = (int)((blockIdx.x * blockDim.x + threadIdx.x) >> 5);
    const int lane = threadIdx.x & 31;
    if (gw >= N_SAMP * C_CH) return;
    const float* p = feats + (size_t)gw * HW;
    float s = 0.0f;
    for (int i = lane; i < HW; i += 32) s += p[i];
#pragma unroll
    for (int off = 16; off > 0; off >>= 1) s += __shfl_xor(s, off, 32);
    if (lane == 0) pooledh[gw] = (_Float16)(s * (1.0f / (float)HW));
}

// ---------------------------------------------------------------------------
// Kernel 2: channel sums.  Thread t (<196) owns spatial position t; per
// channel iteration the wave reads a contiguous 784B row (coalesced).
// global_prefetch hint keeps the stream ahead of the strided walk.
// ---------------------------------------------------------------------------
__global__ __launch_bounds__(224) void chansum_kernel(const float* __restrict__ x,
                                                      float* __restrict__ out) {
    const int n = blockIdx.x;
    const int t = threadIdx.x;
    if (t >= HW) return;
    const float* p = x + (size_t)n * C_CH * HW + t;
    float s = 0.0f;
    for (int c = 0; c < C_CH; ++c) {
        __builtin_prefetch(p + (size_t)(c + 32) * HW, 0, 0); // global_prefetch_b8
        s += p[(size_t)c * HW];
    }
    out[n * HW + t] = s;
}

// ---------------------------------------------------------------------------
// Kernel 3: L1 distances to the 7 centroid maps + argmin.  One wave per n.
// ---------------------------------------------------------------------------
__global__ __launch_bounds__(32) void argmin_kernel(const float* __restrict__ a,
                                                    const float* __restrict__ ks,
                                                    int* __restrict__ labels) {
    const int n    = blockIdx.x;
    const int lane = threadIdx.x;
    float d[K_CLS];
#pragma unroll
    for (int k = 0; k < K_CLS; ++k) d[k] = 0.0f;
    for (int hw = lane; hw < HW; hw += 32) {
        const float av = a[n * HW + hw];
#pragma unroll
        for (int k = 0; k < K_CLS; ++k) d[k] += fabsf(av - ks[k * HW + hw]);
    }
#pragma unroll
    for (int k = 0; k < K_CLS; ++k)
#pragma unroll
        for (int off = 16; off > 0; off >>= 1) d[k] += __shfl_xor(d[k], off, 32);
    if (lane == 0) {
        int best = 0;
        float bv = d[0];
#pragma unroll
        for (int k = 1; k < K_CLS; ++k)
            if (d[k] < bv) { bv = d[k]; best = k; }
        labels[n] = best;
    }
}

// ---------------------------------------------------------------------------
// Kernel 4: selector GEMM + bias + tanh using V_WMMA_F32_16X16X32_F16.
//   sel[n][c] = tanh( b[c] + sum_j pooled[n][j] * Wsel[c][j] )
// Block = 256 threads (8 waves) -> 16(n) x 128(c) tile; K staged 64-wide in
// LDS via async global->LDS b128 copies (ASYNCcnt), 2 WMMAs per stage/wave.
// LDS rows padded to 72 halves (144B, 16B-aligned) for bank spread.
// ---------------------------------------------------------------------------
#define GT_M  16
#define GT_N  128
#define GT_K  64
#define LDP   (GT_K + 8)

__global__ __launch_bounds__(256) void selector_gemm(const _Float16* __restrict__ pooledh,
                                                     const _Float16* __restrict__ Wselh,
                                                     const float* __restrict__ bsel,
                                                     float* __restrict__ sel) {
    __shared__ __attribute__((aligned(16))) _Float16 Alds[GT_M][LDP];
    __shared__ __attribute__((aligned(16))) _Float16 Blds[GT_N][LDP];

    const int tid  = threadIdx.x;
    const int wave = tid >> 5;        // 0..7 -> c sub-tile
    const int lane = tid & 31;
    const int hl   = lane >> 4;       // half-wave select (0 or 1)
    const int row  = lane & 15;       // M (A) / N (B) index within tile

    const int n0 = blockIdx.x * GT_M;
    const int c0 = blockIdx.y * GT_N;

    v8f acc = {};

    for (int k0 = 0; k0 < C_CH; k0 += GT_K) {
        // Stage A (16x64 halves): 16 rows * 8 x 16B segments = 128 transfers.
        if (tid < GT_M * (GT_K / 8)) {
            const int r = tid >> 3, sg = tid & 7;
            async_copy16(pooledh + (size_t)(n0 + r) * C_CH + k0 + sg * 8,
                         &Alds[r][sg * 8]);
        }
        // Stage B (128x64 halves): 1024 transfers, 4 per thread.
        for (int i = tid; i < GT_N * (GT_K / 8); i += 256) {
            const int r = i >> 3, sg = i & 7;
            async_copy16(Wselh + (size_t)(c0 + r) * C_CH + k0 + sg * 8,
                         &Blds[r][sg * 8]);
        }
        async_copy_wait();
        __syncthreads();

#pragma unroll
        for (int ks = 0; ks < GT_K; ks += 32) {
            // A fragment (16x32 f16, ISA 7.12.2): lane holds M=row,
            // K = {8*hl .. 8*hl+7} and {16+8*hl .. 16+8*hl+7}.
            const _Float16* arow = &Alds[row][ks];
            v8h a_lo = *(const v8h*)(arow + hl * 8);
            v8h a_hi = *(const v8h*)(arow + 16 + hl * 8);
            v16h afrag = __builtin_shufflevector(a_lo, a_hi,
                0,1,2,3,4,5,6,7,8,9,10,11,12,13,14,15);

            // B fragment (32x16 f16): lane holds N=row,
            // K = {8*hl .. 8*hl+7} and {16+8*hl .. 16+8*hl+7}.
            const _Float16* brow = &Blds[wave * 16 + row][ks];
            v8h b_lo = *(const v8h*)(brow + hl * 8);
            v8h b_hi = *(const v8h*)(brow + 16 + hl * 8);
            v16h bfrag = __builtin_shufflevector(b_lo, b_hi,
                0,1,2,3,4,5,6,7,8,9,10,11,12,13,14,15);

            acc = __builtin_amdgcn_wmma_f32_16x16x32_f16(
                /*neg_a=*/false, afrag, /*neg_b=*/false, bfrag,
                /*c_mod=*/(short)0, acc, /*reuse_a=*/false, /*reuse_b=*/false);
        }
        __syncthreads();
    }

    // Epilogue: C/D layout — VGPR r: M = r + 8*hl, N = row.
    const int c = c0 + wave * 16 + row;
    const float bias = bsel[c];
#pragma unroll
    for (int r = 0; r < 8; ++r) {
        const int n = n0 + r + 8 * hl;
        sel[(size_t)n * C_CH + c] = tanhf(acc[r] + bias);
    }
}

// ---------------------------------------------------------------------------
// Kernel 5: fused output (bandwidth-dominant pass, ~830MB of traffic):
//   out[n][c][hw] = feats[n][c][hw] + sel[n][c] * centroids[labels[n]][c][hw]
// float4-vectorized, fully coalesced (196 floats = 49 float4 per (n,c)).
// ---------------------------------------------------------------------------
__global__ __launch_bounds__(256) void fuse_out_kernel(const float* __restrict__ feats,
                                                       const float* __restrict__ cents,
                                                       const float* __restrict__ sel,
                                                       const int* __restrict__ labels,
                                                       float* __restrict__ out) {
    const size_t idx = (size_t)blockIdx.x * blockDim.x + threadIdx.x; // float4 units
    const size_t total = (size_t)N_SAMP * C_CH * HW4;
    if (idx >= total) return;
    const int    q  = (int)(idx % HW4);
    const size_t nc = idx / HW4;
    const int    c  = (int)(nc % C_CH);
    const int    n  = (int)(nc / C_CH);

    const int   lbl = labels[n];
    const float s   = sel[nc];

    const float4 f = ((const float4*)feats)[idx];
    const float4 m = ((const float4*)cents)[((size_t)lbl * C_CH + c) * HW4 + q];
    float4 o;
    o.x = fmaf(s, m.x, f.x);
    o.y = fmaf(s, m.y, f.y);
    o.z = fmaf(s, m.z, f.z);
    o.w = fmaf(s, m.w, f.w);
    ((float4*)out)[idx] = o;
}

// ---------------------------------------------------------------------------
// Host-side launch (graph-capture safe: kernels only, all on `stream`).
// ---------------------------------------------------------------------------
extern "C" void kernel_launch(void* const* d_in, const int* in_sizes, int n_in,
                              void* d_out, int out_size, void* d_ws, size_t ws_size,
                              hipStream_t stream) {
    const float* feats = (const float*)d_in[0];   // [256,2048,14,14]
    const float* cents = (const float*)d_in[1];   // [7,2048,14,14]
    const float* Wsel  = (const float*)d_in[2];   // [2048,2048]
    const float* bsel  = (const float*)d_in[3];   // [2048]
    float*       out   = (float*)d_out;

    // Workspace carve-up (~11.5 MB, all offsets 16B-aligned)
    _Float16* Wselh   = (_Float16*)d_ws;                     // 2048*2048 halves
    _Float16* pooledh = Wselh + (size_t)C_CH * C_CH;         // 256*2048 halves
    float*    a_sums  = (float*)(pooledh + (size_t)N_SAMP * C_CH);  // 256*196
    float*    k_sums  = a_sums + (size_t)N_SAMP * HW;        // 7*196
    int*      labels  = (int*)(k_sums + (size_t)K_CLS * HW); // 256
    float*    sel     = (float*)(labels + N_SAMP);           // 256*2048

    // 0) W_sel f32 -> f16 (one-time, L2-resident)
    {
        const int n8 = C_CH * C_CH / 8;
        cvt_f16_kernel<<<(n8 + 255) / 256, 256, 0, stream>>>(Wsel, Wselh, n8);
    }
    // 1) pooled = mean over spatial (f16): one wave per (n,c)
    {
        const int waves  = N_SAMP * C_CH;
        const int blocks = (waves * 32 + 255) / 256;
        pool_kernel<<<blocks, 256, 0, stream>>>(feats, pooledh);
    }
    // 2) channel sums for feats and centroids
    chansum_kernel<<<N_SAMP, 224, 0, stream>>>(feats, a_sums);
    chansum_kernel<<<K_CLS,  224, 0, stream>>>(cents, k_sums);
    // 3) labels = argmin_k sum|a - k|
    argmin_kernel<<<N_SAMP, 32, 0, stream>>>(a_sums, k_sums, labels);
    // 4) selector = tanh(pooled @ Wsel^T + b) via async-LDS + WMMA
    {
        dim3 grid(N_SAMP / GT_M, C_CH / GT_N); // 16 x 16
        selector_gemm<<<grid, 256, 0, stream>>>(pooledh, Wselh, bsel, sel);
    }
    // 5) out = feats + sel * centroids[labels]
    {
        const size_t total  = (size_t)N_SAMP * C_CH * HW4;
        const int    blocks = (int)((total + 255) / 256);
        fuse_out_kernel<<<blocks, 256, 0, stream>>>(feats, cents, sel, labels, out);
    }
}
